// BinaryLinear_81063212745346
// MI455X (gfx1250) — 8-bit matrix operand order repaired and checked by reading, compiled, not run on hardware
//
#include <hip/hip_runtime.h>
#include <stdint.h>

typedef int   v16i __attribute__((ext_vector_type(16)));
typedef float v8f  __attribute__((ext_vector_type(8)));

#define K_DIM 4096
#define N_DIM 4096
#define M_DIM 8192

#define LDS_ROW   144
#define LDS_BUF   (64 * LDS_ROW)

__device__ __forceinline__ unsigned pack4(float a, float b, float c, float d) {
    unsigned b0 = (a >= 0.0f) ? 0x38u : 0xB8u;
    unsigned b1 = (b >= 0.0f) ? 0x38u : 0xB8u;
    unsigned b2 = (c >= 0.0f) ? 0x38u : 0xB8u;
    unsigned b3 = (d >= 0.0f) ? 0x38u : 0xB8u;
    return b0 | (b1 << 8) | (b2 << 16) | (b3 << 24);
}

__global__ __launch_bounds__(256) void binarize_pack_kernel(
    const float* __restrict__ in, uint8_t* __restrict__ out) {
    size_t i = (size_t)blockIdx.x * blockDim.x + threadIdx.x;
    const float4* p = (const float4*)in + i * 4;
    float4 f0 = p[0], f1 = p[1], f2 = p[2], f3 = p[3];
    uint4 r;
    r.x = pack4(f0.x, f0.y, f0.z, f0.w);
    r.y = pack4(f1.x, f1.y, f1.z, f1.w);
    r.z = pack4(f2.x, f2.y, f2.z, f2.w);
    r.w = pack4(f3.x, f3.y, f3.z, f3.w);
    ((uint4*)out)[i] = r;
}

__device__ __forceinline__ void async_copy_B(const uint8_t* __restrict__ gsrc,
                                             uint8_t* lbase, int tid) {
#pragma unroll
    for (int j = 0; j < 2; ++j) {
        int idx = tid + j * 256;
        int r = idx >> 3;
        int s = idx & 7;
        const uint8_t* g = gsrc + (size_t)r * K_DIM + s * 16;
        uint8_t* lptr = lbase + r * LDS_ROW + s * 16;
        unsigned l = (unsigned)(uintptr_t)lptr;
        asm volatile("global_load_async_to_lds_b128 %0, %1, off"
                     :: "v"(l), "v"(g) : "memory");
    }
}

__device__ __forceinline__ void load_B(v16i& B, const uint8_t* sb, int t,
                                       int ncol, int half) {
#pragma unroll
    for (int g = 0; g < 8; ++g) {
        uint2 q = *(const uint2*)(sb + (t * 16 + ncol) * LDS_ROW
                                     + half * 8 + g * 16);
        B[2 * g]     = (int)q.x;
        B[2 * g + 1] = (int)q.y;
    }
}

__device__ __forceinline__ void k_step(const uint8_t* arow0, const uint8_t* arow1,
                                       const uint8_t* sb, int kb,
                                       int ncol, int half, v8f acc[2][4]) {
    v16i A0, A1;
#pragma unroll
    for (int j = 0; j < 4; ++j) {
        uint2 lo0 = *(const uint2*)(arow0 + kb + j * 16);
        uint2 hi0 = *(const uint2*)(arow0 + kb + 64 + j * 16);
        uint2 lo1 = *(const uint2*)(arow1 + kb + j * 16);
        uint2 hi1 = *(const uint2*)(arow1 + kb + 64 + j * 16);
        A0[2 * j] = (int)lo0.x;  A0[2 * j + 1] = (int)lo0.y;
        A0[8 + 2 * j] = (int)hi0.x;  A0[8 + 2 * j + 1] = (int)hi0.y;
        A1[2 * j] = (int)lo1.x;  A1[2 * j + 1] = (int)lo1.y;
        A1[8 + 2 * j] = (int)hi1.x;  A1[8 + 2 * j + 1] = (int)hi1.y;
    }

    v16i Breg[2];
    load_B(Breg[0], sb, 0, ncol, half);
#pragma unroll
    for (int t = 0; t < 4; ++t) {
        if (t < 3) load_B(Breg[(t + 1) & 1], sb, t + 1, ncol, half);
        acc[0][t] = __builtin_amdgcn_wmma_f32_16x16x128_fp8_fp8(
            A0, Breg[t & 1], (short)0, acc[0][t], false, false);
        acc[1][t] = __builtin_amdgcn_wmma_f32_16x16x128_fp8_fp8(
            A1, Breg[t & 1], (short)0, acc[1][t], false, false);
    }
}

__global__ __launch_bounds__(256) void wmma_gemm_kernel(
    const uint8_t* __restrict__ xq,
    const uint8_t* __restrict__ wq,
    const float*   __restrict__ bias,
    float*         __restrict__ out)
{
    __shared__ uint8_t smem[2 * LDS_BUF];

    const int tid  = threadIdx.x;
    const int lane = tid & 31;
    const int w    = tid >> 5;

    const int bm = blockIdx.x >> 6;
    const int bn = blockIdx.x & 63;
    const int mblock = bm * 256;
    const int nblock = bn * 64;

    const int mrowsel = lane & 15;
    const int half    = lane >> 4;
    const int ncol    = lane & 15;

    const uint8_t* arow0 = xq + (size_t)(mblock + w * 32 + mrowsel) * K_DIM + half * 8;
    const uint8_t* arow1 = arow0 + (size_t)16 * K_DIM;
    const uint8_t* bsrc  = wq + (size_t)nblock * K_DIM;

    v8f acc[2][4];
#pragma unroll
    for (int mt = 0; mt < 2; ++mt)
#pragma unroll
        for (int t = 0; t < 4; ++t)
#pragma unroll
            for (int r = 0; r < 8; ++r) acc[mt][t][r] = 0.0f;

    async_copy_B(bsrc, smem, tid);

    int kb = 0;
    for (; kb < K_DIM - 128; kb += 128) {
        const unsigned cur = ((unsigned)kb >> 7) & 1u;
        async_copy_B(bsrc + kb + 128, smem + (cur ^ 1u) * LDS_BUF, tid);
        asm volatile("s_wait_asynccnt 0x2" ::: "memory");
        __syncthreads();
        __builtin_prefetch(arow0 + kb + 128, 0, 3);
        __builtin_prefetch(arow1 + kb + 128, 0, 3);
        k_step(arow0, arow1, smem + cur * LDS_BUF, kb, ncol, half, acc);
        __syncthreads();
    }
    asm volatile("s_wait_asynccnt 0x0" ::: "memory");
    __syncthreads();
    k_step(arow0, arow1, smem + (((unsigned)kb >> 7) & 1u) * LDS_BUF,
           kb, ncol, half, acc);

#pragma unroll
    for (int mt = 0; mt < 2; ++mt) {
        const int mrow = mblock + w * 32 + mt * 16 + half * 8;
#pragma unroll
        for (int t = 0; t < 4; ++t) {
            const int n = nblock + t * 16 + ncol;
            const float bv = bias[n];
            float* op = out + (size_t)mrow * N_DIM + n;
#pragma unroll
            for (int r = 0; r < 8; ++r)
                op[(size_t)r * N_DIM] = acc[mt][t][r] + bv;
        }
    }
}

extern "C" void kernel_launch(void* const* d_in, const int* in_sizes, int n_in,
                              void* d_out, int out_size, void* d_ws, size_t ws_size,
                              hipStream_t stream) {
    const float* x    = (const float*)d_in[0];
    const float* w    = (const float*)d_in[1];
    const float* bias = (const float*)d_in[2];
    float* out        = (float*)d_out;

    uint8_t* xq = (uint8_t*)d_ws;
    uint8_t* wq = xq + (size_t)M_DIM * K_DIM;

    binarize_pack_kernel<<<8192, 256, 0, stream>>>(x, xq);
    binarize_pack_kernel<<<4096, 256, 0, stream>>>(w, wq);

    wmma_gemm_kernel<<<2048, 256, 0, stream>>>(xq, wq, bias, out);
}
